// MoCo_4810363372846
// MI455X (gfx1250) — matrix operand order repaired and checked by reading, not compiled, not run on hardware
//
#include <hip/hip_runtime.h>
#include <hip/hip_bf16.h>

#define BB   16384
#define DMLP 2048
#define DDIM 128
#define KQ   16384
#define NNUM 64

typedef __bf16 v16bf __attribute__((ext_vector_type(16)));
typedef __bf16 v8bf  __attribute__((ext_vector_type(8)));
typedef float  v8f   __attribute__((ext_vector_type(8)));

#if defined(__gfx1250__) && __has_builtin(__builtin_amdgcn_global_load_async_to_lds_b128) && __has_builtin(__builtin_amdgcn_s_wait_asynccnt)
#define HAS_ASYNC_LDS 1
typedef int vi4 __attribute__((vector_size(16)));            // GCC vector: matches builtin's V4i
typedef __attribute__((address_space(1))) vi4* g4p;          // global int4*
typedef __attribute__((address_space(3))) vi4* l4p;          // LDS int4*
#else
#define HAS_ASYNC_LDS 0
#endif

__device__ __forceinline__ unsigned short f2bf(float f) {
    unsigned u = __float_as_uint(f);
    u += 0x7FFFu + ((u >> 16) & 1u);          // round-to-nearest-even
    return (unsigned short)(u >> 16);
}
__device__ __forceinline__ unsigned pack2(float a, float b) {
    return (unsigned)f2bf(a) | ((unsigned)f2bf(b) << 16);
}
__device__ __forceinline__ float wred(float x) {
#pragma unroll
    for (int m = 16; m > 0; m >>= 1) x += __shfl_xor(x, m, 32);
    return x;
}

// ---------------------------------------------------------------------------
// Elementwise fp32 -> bf16 (row-major), 8 elems/thread.
// ---------------------------------------------------------------------------
__global__ __launch_bounds__(256) void cvt_bf16_kernel(
    const float* __restrict__ X, unsigned short* __restrict__ Y)
{
    const size_t i = ((size_t)blockIdx.x * 256 + threadIdx.x) * 8;
    const float4 a = *(const float4*)(X + i);
    const float4 b = *(const float4*)(X + i + 4);
    uint4 o;
    o.x = pack2(a.x, a.y); o.y = pack2(a.z, a.w);
    o.z = pack2(b.x, b.y); o.w = pack2(b.z, b.w);
    *(uint4*)(Y + i) = o;
}

// ---------------------------------------------------------------------------
// fp32 KxN -> bf16 NxK transpose (32x32 LDS tiles).
// ---------------------------------------------------------------------------
__global__ __launch_bounds__(256) void cvt_transpose_kernel(
    const float* __restrict__ X, unsigned short* __restrict__ Xt, int Kd, int N)
{
    __shared__ float tile[32][33];
    const int k0 = blockIdx.y * 32, n0 = blockIdx.x * 32;
    const int r = threadIdx.x >> 3, c4 = (threadIdx.x & 7) << 2;
    const float4 v = *(const float4*)(X + (size_t)(k0 + r) * N + n0 + c4);
    tile[r][c4 + 0] = v.x; tile[r][c4 + 1] = v.y;
    tile[r][c4 + 2] = v.z; tile[r][c4 + 3] = v.w;
    __syncthreads();
    uint2 o;
    o.x = pack2(tile[c4 + 0][r], tile[c4 + 1][r]);
    o.y = pack2(tile[c4 + 2][r], tile[c4 + 3][r]);
    *(uint2*)(Xt + (size_t)(n0 + r) * Kd + k0 + c4) = o;
}

// ---------------------------------------------------------------------------
// All-bf16 tiled GEMM: C[M,N] = act(A @ Bt^T + bias), fp32 WMMA accum.
// A: MxK bf16 row-major. Bt: NxK bf16 row-major (pre-transposed weights).
// Block tile 128x128, K-step 32, 8 waves (2x4), wave = 64x32 = 4x2 WMMA accs.
// Staging: async global->LDS copies (ASYNCcnt) when available.
// LDS rows padded to 40 elems (80B) so fragment chunks stay 16B-aligned.
// ---------------------------------------------------------------------------
template <bool RELU, bool OUT_BF16>
__global__ __launch_bounds__(256) void gemm_bf16(
    const unsigned short* __restrict__ A, const unsigned short* __restrict__ Bt,
    const float* __restrict__ bias, void* __restrict__ Cp,
    int M, int N, int Kd)
{
    constexpr int LDT = 40;
    __shared__ __align__(16) unsigned short sA[2][128 * LDT];
    __shared__ __align__(16) unsigned short sB[2][128 * LDT];

    const int tid  = threadIdx.x;
    const int lane = tid & 31;
    const int wv   = tid >> 5;
    const int wm   = wv >> 2;          // 0..1
    const int wn   = wv & 3;           // 0..3
    const int half = lane >> 4;
    const int l16  = lane & 15;
    const int bm = blockIdx.y, bn = blockIdx.x;

    // Staging assignment: 2 threads per row, 32B (16 bf16) each.
    const int crow   = tid >> 1;
    const int cchunk = (tid & 1) << 4;
    const unsigned short* gA = A  + (size_t)(bm * 128 + crow) * Kd + cchunk;
    const unsigned short* gB = Bt + (size_t)(bn * 128 + crow) * Kd + cchunk;
    const int ldsOff = crow * LDT + cchunk;

#if HAS_ASYNC_LDS
    auto stage = [&](int k0, int buf) {
        g4p pa = (g4p)(gA + k0);
        g4p pb = (g4p)(gB + k0);
        l4p la = (l4p)&sA[buf][ldsOff];
        l4p lb = (l4p)&sB[buf][ldsOff];
        __builtin_amdgcn_global_load_async_to_lds_b128(pa, la, 0,  0);
        __builtin_amdgcn_global_load_async_to_lds_b128(pa, la, 16, 0);
        __builtin_amdgcn_global_load_async_to_lds_b128(pb, lb, 0,  0);
        __builtin_amdgcn_global_load_async_to_lds_b128(pb, lb, 16, 0);
    };
#else
    uint4 ra[2], rb[2];
    auto fetch = [&](int k0) {
        const uint4* pa = (const uint4*)(gA + k0);
        ra[0] = pa[0]; ra[1] = pa[1];
        const uint4* pb = (const uint4*)(gB + k0);
        rb[0] = pb[0]; rb[1] = pb[1];
    };
    auto commit = [&](int buf) {
        uint4* da = (uint4*)&sA[buf][ldsOff]; da[0] = ra[0]; da[1] = ra[1];
        uint4* db = (uint4*)&sB[buf][ldsOff]; db[0] = rb[0]; db[1] = rb[1];
    };
#endif

    const v8f zero = {0.f, 0.f, 0.f, 0.f, 0.f, 0.f, 0.f, 0.f};
    v8f acc[4][2];
#pragma unroll
    for (int i = 0; i < 4; ++i) { acc[i][0] = zero; acc[i][1] = zero; }

    const int KT = Kd >> 5;
#if HAS_ASYNC_LDS
    stage(0, 0);
    __builtin_amdgcn_s_wait_asynccnt(0);
#else
    fetch(0); commit(0);
#endif
    __syncthreads();

#pragma unroll 1
    for (int kt = 0; kt < KT; ++kt) {
        const int  cur  = kt & 1;
        const bool more = (kt + 1) < KT;
#if HAS_ASYNC_LDS
        if (more) stage((kt + 1) << 5, cur ^ 1);
#else
        if (more) fetch((kt + 1) << 5);
#endif

        v16bf afr[4], bfr[2];
#pragma unroll
        for (int mt = 0; mt < 4; ++mt) {
            // A 16x32 in this code's contiguous k order (B matches): lanes 0-15 hold K0-15, lanes 16-31 K16-31
            const unsigned short* p = &sA[cur][(wm * 64 + mt * 16 + l16) * LDT + half * 16];
            v8bf lo = *(const v8bf*)p;
            v8bf hi = *(const v8bf*)(p + 8);
            afr[mt] = __builtin_shufflevector(lo, hi,
                0, 1, 2, 3, 4, 5, 6, 7, 8, 9, 10, 11, 12, 13, 14, 15);
        }
#pragma unroll
        for (int nt = 0; nt < 2; ++nt) {
            // B 32x16 in this code's contiguous k order (matches A): lane = column, lanes 0-15 K0-15, lanes 16-31 K16-31
            const unsigned short* p = &sB[cur][(wn * 32 + nt * 16 + l16) * LDT + half * 16];
            bfr[nt] = *(const v16bf*)p;
        }
#pragma unroll
        for (int mt = 0; mt < 4; ++mt)
#pragma unroll
            for (int nt = 0; nt < 2; ++nt)
                acc[mt][nt] = __builtin_amdgcn_wmma_f32_16x16x32_bf16(
                    false, afr[mt], false, bfr[nt], (short)0, acc[mt][nt], false, false);

#if HAS_ASYNC_LDS
        if (more) __builtin_amdgcn_s_wait_asynccnt(0);
#else
        if (more) commit(cur ^ 1);
#endif
        __syncthreads();
    }

    unsigned short* Ch = (unsigned short*)Cp;
    float*          Cf = (float*)Cp;
#pragma unroll
    for (int mt = 0; mt < 4; ++mt)
#pragma unroll
        for (int nt = 0; nt < 2; ++nt) {
            const int   col = bn * 128 + wn * 32 + nt * 16 + l16;
            const float bv  = bias[col];
#pragma unroll
            for (int g = 0; g < 8; ++g) {
                const int row = bm * 128 + wm * 64 + mt * 16 + half * 8 + g;
                float v = acc[mt][nt][g] + bv;
                if constexpr (RELU) v = fmaxf(v, 0.f);
                if constexpr (OUT_BF16) Ch[(size_t)row * N + col] = f2bf(v);
                else                    Cf[(size_t)row * N + col] = v;
            }
        }
}

// ---------------------------------------------------------------------------
// Gather qs[n][d] = queue[d, sample_idx[n]] and c[n] = qs_n @ invD @ qs_n.
// ---------------------------------------------------------------------------
__global__ __launch_bounds__(256) void prep_qs(
    const float* __restrict__ queue, const int* __restrict__ sidx,
    const float* __restrict__ invD, float* __restrict__ qs, float* __restrict__ cvec)
{
    __shared__ __align__(16) float s_qs[NNUM * DDIM];
    const int tid = threadIdx.x;
    for (int i = tid; i < NNUM * DDIM; i += 256) {
        const int n = i >> 7, d = i & 127;
        const float v = queue[(size_t)d * KQ + sidx[n]];
        s_qs[i] = v; qs[i] = v;
    }
    __syncthreads();
    const int lane = tid & 31, wv = tid >> 5;
    for (int n = wv; n < NNUM; n += 8) {
        float t = 0.f;
        for (int d = lane; d < DDIM; d += 32) {
            float w = 0.f;
#pragma unroll 4
            for (int e = 0; e < DDIM; ++e) w += invD[d * DDIM + e] * s_qs[n * DDIM + e];
            t += s_qs[n * DDIM + d] * w;
        }
        t = wred(t);
        if (lane == 0) cvec[n] = t;
    }
}

// ---------------------------------------------------------------------------
// Per row: normalize q; dist = mean_n sqrt(qMq - 2 (qM).qs_n + c_n).
// One wave per row, 8 rows per block.
// ---------------------------------------------------------------------------
__global__ __launch_bounds__(256) void dist_kernel(
    const float* __restrict__ Q, const float* __restrict__ invD,
    const float* __restrict__ qs, const float* __restrict__ cvec,
    float* __restrict__ dist)
{
    __shared__ __align__(16) float s_qs[NNUM * DDIM];
    __shared__ float s_c[NNUM];
    __shared__ float s_qn[8][DDIM];
    const int tid = threadIdx.x;
    for (int i = tid; i < NNUM * DDIM; i += 256) s_qs[i] = qs[i];
    if (tid < NNUM) s_c[tid] = cvec[tid];
    __syncthreads();

    const int lane = tid & 31, wv = tid >> 5;
    const int row = blockIdx.x * 8 + wv;

    float4 q = *(const float4*)(Q + (size_t)row * DDIM + lane * 4);
    const float ss = wred(q.x * q.x + q.y * q.y + q.z * q.z + q.w * q.w);
    const float rn = 1.f / fmaxf(sqrtf(ss), 1e-12f);
    q.x *= rn; q.y *= rn; q.z *= rn; q.w *= rn;
    s_qn[wv][lane * 4 + 0] = q.x; s_qn[wv][lane * 4 + 1] = q.y;
    s_qn[wv][lane * 4 + 2] = q.z; s_qn[wv][lane * 4 + 3] = q.w;
    __syncthreads();

    float4 v = {0.f, 0.f, 0.f, 0.f};
    for (int d = 0; d < DDIM; ++d) {
        const float  qd = s_qn[wv][d];
        const float4 r  = *(const float4*)(invD + (size_t)d * DDIM + lane * 4);
        v.x += qd * r.x; v.y += qd * r.y; v.z += qd * r.z; v.w += qd * r.w;
    }
    const float qmq = wred(v.x * q.x + v.y * q.y + v.z * q.z + v.w * q.w);

    float s = 0.f;
    for (int n = 0; n < NNUM; ++n) {
        const float4 g  = *(const float4*)(&s_qs[n * DDIM + lane * 4]);
        const float  cr = wred(v.x * g.x + v.y * g.y + v.z * g.z + v.w * g.w);
        s += sqrtf(fmaxf(qmq - 2.f * cr + s_c[n], 0.f));
    }
    if (lane == 0) dist[row] = s * (1.f / NNUM);
}

// ---------------------------------------------------------------------------
// Stable-argsort rank counting (dist vector in LDS), positions [B-64, B-2],
// fused conditional output rewrite.
// ---------------------------------------------------------------------------
__global__ __launch_bounds__(256) void select_kernel(
    const float* __restrict__ dist, const float* __restrict__ output,
    float* __restrict__ out)
{
    __shared__ float sd[BB];   // 64 KB of the 320 KB WGP LDS
    const int tid = threadIdx.x;
    for (int i = tid; i < BB; i += 256) sd[i] = dist[i];
    __syncthreads();

    const int   r  = blockIdx.x * 256 + tid;
    const float dr = sd[r];
    int pos = 0;
#pragma unroll 4
    for (int j = 0; j < BB; ++j) {
        const float dj = sd[j];
        pos += (dj < dr) ? 1 : ((dj == dr && j < r) ? 1 : 0);
    }
    const bool sel = (pos >= BB - NNUM) && (pos < BB - 1);

    float4 o = *(const float4*)(output + (size_t)r * 4);
    if (sel && (fabsf(o.z) < 1.f || fabsf(o.w) < 1.f)) { o.z = -5.f; o.w = 5.f; }
    *(float4*)(out + (size_t)r * 4) = o;
}

// ---------------------------------------------------------------------------
extern "C" void kernel_launch(void* const* d_in, const int* in_sizes, int n_in,
                              void* d_out, int out_size, void* d_ws, size_t ws_size,
                              hipStream_t stream)
{
    const float* im_q   = (const float*)d_in[0];
    const float* output = (const float*)d_in[1];
    const int*   sidx   = (const int*)d_in[2];
    const float* W1     = (const float*)d_in[3];
    const float* b1     = (const float*)d_in[4];
    const float* W2     = (const float*)d_in[5];
    const float* b2     = (const float*)d_in[6];
    const float* queue  = (const float*)d_in[7];
    const float* invD   = (const float*)d_in[8];

    char* ws = (char*)d_ws;
    size_t off = 0;
    unsigned short* imq16 = (unsigned short*)(ws + off); off += (size_t)BB * DMLP * 2;
    unsigned short* W1t   = (unsigned short*)(ws + off); off += (size_t)DMLP * DMLP * 2;
    unsigned short* W2t   = (unsigned short*)(ws + off); off += (size_t)DDIM * DMLP * 2;
    unsigned short* H     = (unsigned short*)(ws + off); off += (size_t)BB * DMLP * 2;
    float* Q    = (float*)(ws + off); off += (size_t)BB * DDIM * 4;
    float* qs   = (float*)(ws + off); off += (size_t)NNUM * DDIM * 4;
    float* cvec = (float*)(ws + off); off += 256;
    float* dist = (float*)(ws + off);

    // One-time precision/layout prep (streaming, ~9 us of HBM time total).
    cvt_bf16_kernel<<<(BB * DMLP) / 2048, 256, 0, stream>>>(im_q, imq16);
    cvt_transpose_kernel<<<dim3(DMLP / 32, DMLP / 32), 256, 0, stream>>>(W1, W1t, DMLP, DMLP);
    cvt_transpose_kernel<<<dim3(DDIM / 32, DMLP / 32), 256, 0, stream>>>(W2, W2t, DMLP, DDIM);
    prep_qs<<<1, 256, 0, stream>>>(queue, sidx, invD, qs, cvec);

    // H = relu(im_q @ W1 + b1)   (bf16 in/out, fp32 accum)
    gemm_bf16<true, true><<<dim3(DMLP / 128, BB / 128), 256, 0, stream>>>(
        imq16, W1t, b1, (void*)H, BB, DMLP, DMLP);

    // Q = H @ W2 + b2            (fp32 out)
    gemm_bf16<false, false><<<dim3(DDIM / 128, BB / 128), 256, 0, stream>>>(
        H, W2t, b2, (void*)Q, BB, DDIM, DMLP);

    dist_kernel<<<BB / 8, 256, 0, stream>>>(Q, invD, qs, cvec, dist);
    select_kernel<<<BB / 256, 256, 0, stream>>>(dist, output, (float*)d_out);
}
